// CausalSelfAttention_5102421147678
// MI455X (gfx1250) — compile-verified, matrix operand order checked by reading, not run on hardware
//
#include <hip/hip_runtime.h>

typedef __attribute__((ext_vector_type(16))) __bf16 v16bf;
typedef __attribute__((ext_vector_type(8)))  __bf16 v8bf;
typedef __attribute__((ext_vector_type(8)))  float  v8f;

#define DEV __device__ __forceinline__

static constexpr int Tt = 2048;     // sequence length
static constexpr int Cc = 1024;     // model dim
static constexpr int Hh = 16;       // heads
static constexpr int Dh = 64;       // head dim
static constexpr int Mtot = 2 * Tt; // B*T = 4096

DEV v16bf pack16(v8bf lo, v8bf hi) {
  v16bf r;
#pragma unroll
  for (int i = 0; i < 8; ++i) { r[i] = lo[i]; r[i + 8] = hi[i]; }
  return r;
}

// A/B fragment chunk (ISA 7.12.2, 16-bit 16x32): lane half h holds k = 8h+i and 16+8h+i.
DEV v16bf load_frag(const __bf16* rowptr, int half) {
  v8bf lo = *(const v8bf*)(rowptr + 8 * half);
  v8bf hi = *(const v8bf*)(rowptr + 16 + 8 * half);
  return pack16(lo, hi);
}

DEV float rowmax16(float x) {
#pragma unroll
  for (int o = 1; o < 16; o <<= 1) x = fmaxf(x, __shfl_xor(x, o, 32));
  return x;
}
DEV float rowsum16(float x) {
#pragma unroll
  for (int o = 1; o < 16; o <<= 1) x += __shfl_xor(x, o, 32);
  return x;
}

DEV uint32_t lds_off(const void* p) {          // flat LDS addr -> 32-bit LDS offset
  return (uint32_t)(uintptr_t)p;
}

// async global (16B) -> LDS copy, tracked with ASYNCcnt
DEV void async_copy_b128(const __bf16* g, const __bf16* l) {
  uint32_t la = lds_off(l);
  uint64_t ga = (uint64_t)(uintptr_t)g;
  asm volatile("global_load_async_to_lds_b128 %0, %1, off" :: "v"(la), "v"(ga) : "memory");
}

// ---------------- conversion kernels ----------------
__global__ void cvt_f32_bf16(const float* __restrict__ in, __bf16* __restrict__ out, int n) {
  int i = blockIdx.x * blockDim.x + threadIdx.x;
  if (i < n) out[i] = (__bf16)in[i];
}

// in: [K, N] f32 row-major  ->  out: [N, K] bf16 row-major
__global__ void transpose_f32_bf16(const float* __restrict__ in, __bf16* __restrict__ out,
                                   int K, int N) {
  int i = blockIdx.x * blockDim.x + threadIdx.x;
  if (i >= K * N) return;
  int n = i / K;
  int k = i - n * K;
  out[i] = (__bf16)in[(size_t)k * N + n];
}

// ---------------- shared GEMM core ----------------
// 256 threads = 8 waves (4 in M x 2 in N). Block tile 128x128, K-chunk 32,
// double-buffered async global->LDS staging. acc[i][c]: i = M-frag (2), c = N-frag (4).
DEV void gemm_core(const __bf16* __restrict__ A, const __bf16* __restrict__ Bt, int Kd,
                   int m0blk, int n0blk, __bf16* sA, __bf16* sB, v8f acc[2][4]) {
  const int tid  = threadIdx.x;
  const int lane = tid & 31;
  const int wid  = tid >> 5;
  const int half = lane >> 4;
  const int ln   = lane & 15;
  const int wm   = wid >> 1;   // 0..3
  const int wn   = wid & 1;    // 0..1

  // stage one 128x32 bf16 tile of A and of Bt into LDS buffer `buf`
  auto issue = [&](int buf, int k0) {
#pragma unroll
    for (int r = 0; r < 2; ++r) {
      int c   = tid + r * 256;       // 512 chunks of 16B per tile
      int row = c >> 2;              // 4 chunks per 32-elem row
      int cc  = c & 3;
      async_copy_b128(A  + (size_t)(m0blk + row) * Kd + k0 + cc * 8,
                      sA + buf * 4096 + row * 32 + cc * 8);
      async_copy_b128(Bt + (size_t)(n0blk + row) * Kd + k0 + cc * 8,
                      sB + buf * 4096 + row * 32 + cc * 8);
    }
  };

  const int nIter = Kd / 32;
  issue(0, 0);
  for (int kc = 0; kc < nIter; ++kc) {
    int buf = kc & 1;
    if (kc + 1 < nIter) {
      issue(buf ^ 1, (kc + 1) * 32);
      asm volatile("s_wait_asynccnt 4" ::: "memory");  // this chunk's 4 copies done
    } else {
      asm volatile("s_wait_asynccnt 0" ::: "memory");
    }
    __syncthreads();

    const __bf16* a0 = sA + buf * 4096;
    const __bf16* b0 = sB + buf * 4096;
    v16bf af[2];
#pragma unroll
    for (int i = 0; i < 2; ++i)
      af[i] = load_frag(a0 + (wm * 32 + i * 16 + ln) * 32, half);
#pragma unroll
    for (int c = 0; c < 4; ++c) {
      v16bf bf = load_frag(b0 + (wn * 64 + c * 16 + ln) * 32, half);
#pragma unroll
      for (int i = 0; i < 2; ++i)
        acc[i][c] = __builtin_amdgcn_wmma_f32_16x16x32_bf16(false, af[i], false, bf,
                                                            (short)0, acc[i][c], false, false);
    }
    __syncthreads();   // everyone done with buf before it is overwritten
  }
}

// ---------------- QKV GEMM: (4096x1024) @ (1024x3072) + bias, scattered ------------
__global__ void gemm_qkv(const __bf16* __restrict__ A, const __bf16* __restrict__ Bt,
                         const float* __restrict__ bias,
                         __bf16* __restrict__ q, __bf16* __restrict__ kmat,
                         __bf16* __restrict__ vt) {
  __shared__ __bf16 sA[2 * 4096];
  __shared__ __bf16 sB[2 * 4096];
  const int lane = threadIdx.x & 31;
  const int wid  = threadIdx.x >> 5;
  const int half = lane >> 4;
  const int ln   = lane & 15;
  const int wm   = wid >> 1, wn = wid & 1;
  const int m0blk = blockIdx.y * 128;
  const int n0blk = blockIdx.x * 128;

  v8f acc[2][4] = {};
  gemm_core(A, Bt, Cc, m0blk, n0blk, sA, sB, acc);

#pragma unroll
  for (int i = 0; i < 2; ++i) {
#pragma unroll
    for (int c = 0; c < 4; ++c) {
      int n = n0blk + wn * 64 + c * 16 + ln;
      float bv = bias[n];
#pragma unroll
      for (int v = 0; v < 8; ++v) {
        int m = m0blk + wm * 32 + i * 16 + v + 8 * half;
        float val = acc[i][c][v] + bv;
        int b = m >> 11;
        int t = m & (Tt - 1);
        if (n < Cc) {
          int h = n >> 6, d = n & 63;
          q[(((size_t)(b * Hh + h)) * Tt + t) * Dh + d] = (__bf16)val;
        } else if (n < 2 * Cc) {
          int nn = n - Cc;
          int h = nn >> 6, d = nn & 63;
          kmat[(((size_t)(b * Hh + h)) * Tt + t) * Dh + d] = (__bf16)val;
        } else {
          int nn = n - 2 * Cc;
          int h = nn >> 6, d = nn & 63;
          vt[(((size_t)(b * Hh + h)) * Dh + d) * Tt + t] = (__bf16)val;  // V transposed
        }
      }
    }
  }
}

// ---------------- projection GEMM: (4096x1024) @ (1024x1024) + bias -> f32 ---------
__global__ void gemm_proj(const __bf16* __restrict__ A, const __bf16* __restrict__ Bt,
                          const float* __restrict__ bias, float* __restrict__ out) {
  __shared__ __bf16 sA[2 * 4096];
  __shared__ __bf16 sB[2 * 4096];
  const int lane = threadIdx.x & 31;
  const int wid  = threadIdx.x >> 5;
  const int half = lane >> 4;
  const int ln   = lane & 15;
  const int wm   = wid >> 1, wn = wid & 1;
  const int m0blk = blockIdx.y * 128;
  const int n0blk = blockIdx.x * 128;

  v8f acc[2][4] = {};
  gemm_core(A, Bt, Cc, m0blk, n0blk, sA, sB, acc);

#pragma unroll
  for (int i = 0; i < 2; ++i) {
#pragma unroll
    for (int c = 0; c < 4; ++c) {
      int n = n0blk + wn * 64 + c * 16 + ln;
      float bv = bias[n];
#pragma unroll
      for (int v = 0; v < 8; ++v) {
        int m = m0blk + wm * 32 + i * 16 + v + 8 * half;
        out[(size_t)m * Cc + n] = acc[i][c][v] + bv;
      }
    }
  }
}

// ---------------- flash attention ----------------
// grid: (T/128, B*H), block 256 (8 waves). Each wave: 16 query rows, online softmax
// over 32-key blocks. Y written bf16 [B*T, C] row-major for the proj GEMM.
__global__ void attn_flash(const __bf16* __restrict__ Q, const __bf16* __restrict__ Km,
                           const __bf16* __restrict__ Vt, __bf16* __restrict__ Y) {
  __shared__ float plds[8][16 * 32];   // per-wave P transpose tile (16 KB)

  int wid  = threadIdx.x >> 5;
  int lane = threadIdx.x & 31;
  int half = lane >> 4;
  int ln   = lane & 15;
  int bh = blockIdx.y;
  int b  = bh / Hh;
  int h  = bh - b * Hh;
  int q0 = blockIdx.x * 128 + wid * 16;

  const __bf16* Qb = Q  + (size_t)bh * Tt * Dh;
  const __bf16* Kb = Km + (size_t)bh * Tt * Dh;
  const __bf16* Vb = Vt + (size_t)bh * Dh * Tt;
  const float scale = 0.125f;  // 1/sqrt(64)
  const float NEG_INF = -__builtin_inff();

  v16bf qf[2];
#pragma unroll
  for (int kc = 0; kc < 2; ++kc)
    qf[kc] = load_frag(Qb + (size_t)(q0 + ln) * Dh + kc * 32, half);

  v8f accO[4] = {};
  float mrow[8], lrow[8];
#pragma unroll
  for (int v = 0; v < 8; ++v) { mrow[v] = NEG_INF; lrow[v] = 0.0f; }

  int nkb = (q0 >> 5) + 1;   // 32-key blocks needed (q0 is a multiple of 16)
  for (int kb = 0; kb < nkb; ++kb) {
    int kbase = kb * 32;
    v8f s0 = {}, s1 = {};
#pragma unroll
    for (int kc = 0; kc < 2; ++kc) {
      v16bf bf0 = load_frag(Kb + (size_t)(kbase + ln) * Dh + kc * 32, half);
      s0 = __builtin_amdgcn_wmma_f32_16x16x32_bf16(false, qf[kc], false, bf0,
                                                   (short)0, s0, false, false);
      v16bf bf1 = load_frag(Kb + (size_t)(kbase + 16 + ln) * Dh + kc * 32, half);
      s1 = __builtin_amdgcn_wmma_f32_16x16x32_bf16(false, qf[kc], false, bf1,
                                                   (short)0, s1, false, false);
    }

    float av[8];
#pragma unroll
    for (int v = 0; v < 8; ++v) {
      int qi = q0 + v + 8 * half;
      float a0 = s0[v] * scale;
      float a1 = s1[v] * scale;
      if (kbase + ln > qi)      a0 = NEG_INF;
      if (kbase + 16 + ln > qi) a1 = NEG_INF;
      float rm = rowmax16(fmaxf(a0, a1));
      float mnew = fmaxf(mrow[v], rm);
      float alpha = __expf(mrow[v] - mnew);
      float p0 = __expf(a0 - mnew);
      float p1 = __expf(a1 - mnew);
      float rs = rowsum16(p0 + p1);
      lrow[v] = lrow[v] * alpha + rs;
      mrow[v] = mnew;
      av[v] = alpha;
      plds[wid][(v + 8 * half) * 32 + ln]      = p0;   // wave-private, DS in-order
      plds[wid][(v + 8 * half) * 32 + 16 + ln] = p1;
    }
#pragma unroll
    for (int c = 0; c < 4; ++c)
#pragma unroll
      for (int v = 0; v < 8; ++v) accO[c][v] *= av[v];

    v16bf pf;   // P as 16x32 bf16 A-fragment, read transposed from LDS
#pragma unroll
    for (int i = 0; i < 8; ++i) {
      pf[i]     = (__bf16)plds[wid][ln * 32 + 8 * half + i];
      pf[i + 8] = (__bf16)plds[wid][ln * 32 + 16 + 8 * half + i];
    }

#pragma unroll
    for (int c = 0; c < 4; ++c) {
      v16bf vf = load_frag(Vb + (size_t)(c * 16 + ln) * Tt + kbase, half);
      accO[c] = __builtin_amdgcn_wmma_f32_16x16x32_bf16(false, pf, false, vf,
                                                        (short)0, accO[c], false, false);
    }
  }

#pragma unroll
  for (int c = 0; c < 4; ++c) {
    int d = c * 16 + ln;
#pragma unroll
    for (int v = 0; v < 8; ++v) {
      int t = q0 + v + 8 * half;
      float o = accO[c][v] / lrow[v];
      Y[((size_t)b * Tt + t) * Cc + h * Dh + d] = (__bf16)o;
    }
  }
}

// ---------------- launch ----------------
extern "C" void kernel_launch(void* const* d_in, const int* in_sizes, int n_in,
                              void* d_out, int out_size, void* d_ws, size_t ws_size,
                              hipStream_t stream) {
  const float* x     = (const float*)d_in[0];   // (2,2048,1024)
  const float* Wqkv  = (const float*)d_in[1];   // (1024,3072)
  const float* bqkv  = (const float*)d_in[2];   // (3072,)
  const float* Wproj = (const float*)d_in[3];   // (1024,1024)
  const float* bproj = (const float*)d_in[4];   // (1024,)

  char* ws = (char*)d_ws;
  const size_t MB = 1u << 20;
  __bf16* xb     = (__bf16*)(ws + 0 * MB);   // 8 MB
  __bf16* wqkvT  = (__bf16*)(ws + 8 * MB);   // 6 MB
  __bf16* wprojT = (__bf16*)(ws + 14 * MB);  // 2 MB
  __bf16* qb     = (__bf16*)(ws + 16 * MB);  // 8 MB  [B,H,T,D]
  __bf16* kb     = (__bf16*)(ws + 24 * MB);  // 8 MB  [B,H,T,D]
  __bf16* vtb    = (__bf16*)(ws + 32 * MB);  // 8 MB  [B,H,D,T]
  __bf16* yb     = (__bf16*)(ws + 40 * MB);  // 8 MB  [B*T, C]   (48 MB total)

  cvt_f32_bf16<<<(Mtot * Cc) / 256, 256, 0, stream>>>(x, xb, Mtot * Cc);
  transpose_f32_bf16<<<(Cc * 3 * Cc) / 256, 256, 0, stream>>>(Wqkv, wqkvT, Cc, 3 * Cc);
  transpose_f32_bf16<<<(Cc * Cc) / 256, 256, 0, stream>>>(Wproj, wprojT, Cc, Cc);

  gemm_qkv<<<dim3((3 * Cc) / 128, Mtot / 128), 256, 0, stream>>>(xb, wqkvT, bqkv,
                                                                 qb, kb, vtb);

  attn_flash<<<dim3(Tt / 128, 2 * Hh), 256, 0, stream>>>(qb, kb, vtb, yb);

  gemm_proj<<<dim3(Cc / 128, Mtot / 128), 256, 0, stream>>>(yb, wprojT, bproj,
                                                            (float*)d_out);
}
